// SinkhornModel_66683662238186
// MI455X (gfx1250) — matrix operand order repaired and checked by reading, not compiled, not run on hardware
//
#include <hip/hip_runtime.h>

#define NR   50512      // rows
#define NC   2048       // centroids
#define EMB  300
#define KPAD 320        // K padded to multiple of 32
#define INV_EPS 100.0f
#define A_CONST ((float)NC / (float)NR)
#define MT    (NR / 16)          // 3157 wave-row tiles
#define MBLK  ((MT + 7) / 8)     // 395 m-blocks of 8 wave tiles

typedef _Float16 v8h  __attribute__((ext_vector_type(8)));
typedef _Float16 v16h __attribute__((ext_vector_type(16)));
typedef float    v8f  __attribute__((ext_vector_type(8)));
typedef float    v4f  __attribute__((ext_vector_type(4)));
typedef unsigned int u32x4 __attribute__((ext_vector_type(4)));
typedef int      i32x4 __attribute__((ext_vector_type(4)));
typedef int      i32x8 __attribute__((ext_vector_type(8)));

__device__ __forceinline__ float warp_sum(float v) {
  #pragma unroll
  for (int m = 16; m > 0; m >>= 1) v += __shfl_xor(v, m, 32);
  return v;
}

// ---------------------------------------------------------------- init
__global__ void init_state(float* v0, float* v1, float* colsum, float* out) {
  int j = blockIdx.x * blockDim.x + threadIdx.x;
  if (j < NC) { v0[j] = 1.0f; v1[j] = 1.0f; colsum[j] = 0.0f; }
  if (j == 0) out[0] = 0.0f;
}

// ------------------------------------------------- row L2-normalize -> f16 (K padded)
__global__ void norm_rows(const float* __restrict__ X, _Float16* __restrict__ Y, int R) {
  const int lane = threadIdx.x & 31;
  const int row  = blockIdx.x * (blockDim.x >> 5) + (threadIdx.x >> 5);
  if (row >= R) return;
  const float* x = X + (size_t)row * EMB;
  float s = 0.0f;
  for (int i = lane; i < EMB; i += 32) { float t = x[i]; s += t * t; }
  s = warp_sum(s);
  s = __shfl(s, 0, 32);
  const float rn = rsqrtf(s);
  _Float16* y = Y + (size_t)row * KPAD;
  for (int i = lane; i < KPAD; i += 32)
    y[i] = (_Float16)(i < EMB ? x[i] * rn : 0.0f);
}

// ------------------------------------------------- GEMM: score = A @ B^T * 100, f16 out
// Block = 8 waves sharing one 128-col B panel staged in LDS by the Tensor Data
// Mover (one tensor_load_to_lds of 80KB). Each wave computes a 16x128 tile with
// v_wmma_f32_16x16x32_f16; A fragments (all 10 k-steps) are preloaded into
// registers so the inner loop is pure ds_load_b128 + WMMA.
__global__ __launch_bounds__(256) void gemm_score(const _Float16* __restrict__ A,
                                                  const _Float16* __restrict__ B,
                                                  _Float16* __restrict__ S) {
  __shared__ _Float16 Bs[128 * KPAD];            // 80 KB B panel

  const int lane  = threadIdx.x & 31;
  const int wave  = threadIdx.x >> 5;
  const int nb    = (blockIdx.x / MBLK) * 128;   // shared by all 8 waves
  const int mblk  = blockIdx.x % MBLK;
  int mtile = mblk * 8 + wave;
  if (mtile > MT - 1) mtile = MT - 1;            // clamp ragged edge (benign dup)
  const int m0  = mtile * 16;
  const int l15 = lane & 15;
  const int hi  = lane >> 4;                     // 0: lanes 0-15, 1: lanes 16-31

  // --- TDM: DMA the B panel (rows nb..nb+127, all K) into LDS --------------
  if (wave == 0) {
    unsigned long long ga = (unsigned long long)(const void*)(B + (size_t)nb * KPAD);
    unsigned ldsoff = (unsigned)(unsigned long long)(&Bs[0]);
    u32x4 g0;
    g0[0] = 1u;                                          // count=1 (valid), user mode
    g0[1] = ldsoff;                                      // lds_addr
    g0[2] = (unsigned)(ga & 0xFFFFFFFFu);                // global_addr[31:0]
    g0[3] = (unsigned)((ga >> 32) & 0x01FFFFFFu) | (2u << 30); // addr[56:32] | type=2
    i32x8 g1 = {};
    g1[0] = (1 << 16);        // data_size = 1 -> 2 bytes/element
    g1[1] = (KPAD << 16);     // tensor_dim0[15:0] = 320
    g1[2] = (128 << 16);      // tensor_dim1[15:0] = 128 rows
    g1[3] = (KPAD << 16);     // tile_dim0 = 320
    g1[4] = 128;              // tile_dim1 = 128, tile_dim2 = 0 (2-D)
    g1[5] = KPAD;             // tensor_dim0_stride[31:0] = 320
    g1[6] = 0;                // stride[47:32]=0, tensor_dim1_stride=0
    g1[7] = 0;
    i32x4 gz4 = {};
    i32x8 gz8 = {};
    __builtin_amdgcn_tensor_load_to_lds(g0, g1, gz4, gz4, gz8, 0);
  }

  // --- Preload all A fragments (10 k-steps) while the DMA runs -------------
  const _Float16* Arow = A + (size_t)(m0 + l15) * KPAD + (hi ? 8 : 0);
  v16h afrag[10];
  #pragma unroll
  for (int kt = 0; kt < 10; ++kt) {
    v8h alo = *(const v8h*)(Arow + kt * 32);
    v8h ahi = *(const v8h*)(Arow + kt * 32 + 16);
    afrag[kt] = __builtin_shufflevector(alo, ahi, 0,1,2,3,4,5,6,7,8,9,10,11,12,13,14,15);
  }

  if (wave == 0) __builtin_amdgcn_s_wait_tensorcnt(0);
  __syncthreads();

  // --- Main loop: ds_load_b128 B fragments + WMMA --------------------------
  v8f acc[8] = {};
  const int kb = hi ? 8 : 0;
  #pragma unroll
  for (int kt = 0; kt < 10; ++kt) {
    const int k0 = kt * 32;
    #pragma unroll
    for (int t = 0; t < 8; ++t) {
      const _Float16* brow = &Bs[(t * 16 + l15) * KPAD + k0 + kb];
      v8h blo = *(const v8h*)brow;
      v8h bhi = *(const v8h*)(brow + 16);
      v16h bf = __builtin_shufflevector(blo, bhi, 0,1,2,3,4,5,6,7,8,9,10,11,12,13,14,15);
      acc[t] = __builtin_amdgcn_wmma_f32_16x16x32_f16(
          false, afrag[kt], false, bf, (short)0, acc[t], false, false);
    }
  }

  // --- Epilogue: C layout (8 VGPRs): n = lane&15, m = r + (hi?8:0) ---------
  #pragma unroll
  for (int t = 0; t < 8; ++t) {
    const int n = nb + t * 16 + l15;
    #pragma unroll
    for (int r = 0; r < 8; ++r) {
      const int m = m0 + r + (hi ? 8 : 0);
      S[(size_t)m * NC + n] = (_Float16)(acc[t][r] * INV_EPS);
    }
  }
}

// ------------------------------------------------- Sinkhorn row pass: u = a / (K @ v)
__global__ __launch_bounds__(256) void sinkhorn_row(const _Float16* __restrict__ S,
                                                    const float* __restrict__ v,
                                                    float* __restrict__ u) {
  const int lane = threadIdx.x & 31;
  const int row  = blockIdx.x * 8 + (threadIdx.x >> 5);
  const _Float16* srow = S + (size_t)row * NC;
  float acc = 0.0f;
  for (int c = lane * 8; c < NC; c += 256) {
    v8h sc = *(const v8h*)(srow + c);
    v4f va = *(const v4f*)(v + c);
    v4f vb = *(const v4f*)(v + c + 4);
    #pragma unroll
    for (int e = 0; e < 4; ++e) acc += __expf((float)sc[e]) * va[e];
    #pragma unroll
    for (int e = 0; e < 4; ++e) acc += __expf((float)sc[4 + e]) * vb[e];
  }
  acc = warp_sum(acc);
  if (lane == 0) u[row] = A_CONST / acc;
}

// ------------------------------------------------- row pass: lse_i = log(sum_j exp(s_ij))
__global__ __launch_bounds__(256) void row_lse(const _Float16* __restrict__ S,
                                               float* __restrict__ lse) {
  const int lane = threadIdx.x & 31;
  const int row  = blockIdx.x * 8 + (threadIdx.x >> 5);
  const _Float16* srow = S + (size_t)row * NC;
  float acc = 0.0f;
  for (int c = lane * 8; c < NC; c += 256) {
    v8h sc = *(const v8h*)(srow + c);
    #pragma unroll
    for (int e = 0; e < 8; ++e) acc += __expf((float)sc[e]);
  }
  acc = warp_sum(acc);
  if (lane == 0) lse[row] = __logf(acc);
}

// ------------------------------------------------- Sinkhorn col pass: colsum_j += u_i K_ij
__global__ __launch_bounds__(256) void sinkhorn_col(const _Float16* __restrict__ S,
                                                    const float* __restrict__ u,
                                                    float* __restrict__ colsum) {
  const int c  = threadIdx.x * 8;        // 256 threads x 8 cols = 2048
  const int r0 = blockIdx.x * 32;
  float acc[8] = {};
  for (int rr = 0; rr < 32; ++rr) {
    const int r = r0 + rr;
    if (r >= NR) break;
    const float ur = u[r];
    const _Float16* srow = S + (size_t)r * NC + c;
    if (r + 1 < NR) __builtin_prefetch(srow + NC, 0, 0);   // global_prefetch_b8
    v8h sc = *(const v8h*)srow;
    #pragma unroll
    for (int e = 0; e < 8; ++e) acc[e] += ur * __expf((float)sc[e]);
  }
  #pragma unroll
  for (int e = 0; e < 8; ++e) atomicAdd(&colsum[c + e], acc[e]);
}

// ------------------------------------------------- v_j = 1/colsum_j ; reset colsum
__global__ void finish_v(float* __restrict__ colsum, float* __restrict__ v) {
  const int j = blockIdx.x * blockDim.x + threadIdx.x;
  if (j < NC) { v[j] = 1.0f / colsum[j]; colsum[j] = 0.0f; }
}

// ------------------------------------------------- fused loss reduction
__global__ __launch_bounds__(256) void final_loss(const _Float16* __restrict__ S0,
                                                  const _Float16* __restrict__ S1,
                                                  const float* __restrict__ u0,
                                                  const float* __restrict__ v0,
                                                  const float* __restrict__ u1,
                                                  const float* __restrict__ v1,
                                                  const float* __restrict__ lse0,
                                                  const float* __restrict__ lse1,
                                                  float* __restrict__ out) {
  const int c  = threadIdx.x * 8;
  const int r0 = blockIdx.x * 32;
  float vv0[8], vv1[8];
  #pragma unroll
  for (int e = 0; e < 8; ++e) { vv0[e] = v0[c + e]; vv1[e] = v1[c + e]; }
  float acc = 0.0f;
  for (int rr = 0; rr < 32; ++rr) {
    const int r = r0 + rr;
    if (r >= NR) break;
    const float u0r = u0[r], u1r = u1[r], l0 = lse0[r], l1 = lse1[r];
    v8h s0 = *(const v8h*)(S0 + (size_t)r * NC + c);
    v8h s1 = *(const v8h*)(S1 + (size_t)r * NC + c);
    #pragma unroll
    for (int e = 0; e < 8; ++e) {
      const float f0 = (float)s0[e], f1 = (float)s1[e];
      acc += u1r * __expf(f1) * vv1[e] * (f0 - l0)    // cpl1 * lp0
           + u0r * __expf(f0) * vv0[e] * (f1 - l1);   // cpl0 * lp1
    }
  }
  acc = warp_sum(acc);
  __shared__ float sm[8];
  const int warp = threadIdx.x >> 5, lane = threadIdx.x & 31;
  if (lane == 0) sm[warp] = acc;
  __syncthreads();
  if (warp == 0) {
    float t = (lane < 8) ? sm[lane] : 0.0f;
    t = warp_sum(t);
    if (lane == 0) atomicAdd(out, -t * (1.0f / (float)NR));
  }
}

// ----------------------------------------------------------------------------
extern "C" void kernel_launch(void* const* d_in, const int* in_sizes, int n_in,
                              void* d_out, int out_size, void* d_ws, size_t ws_size,
                              hipStream_t stream) {
  (void)in_sizes; (void)n_in; (void)out_size; (void)ws_size;
  const float* nf0 = (const float*)d_in[0];
  const float* nf1 = (const float*)d_in[1];
  const float* cen = (const float*)d_in[2];
  float* out = (float*)d_out;

  char* base = (char*)d_ws;
  size_t off = 0;
  auto take = [&](size_t bytes) -> char* {
    off = (off + 255) & ~(size_t)255;
    char* p = base + off;
    off += bytes;
    return p;
  };
  _Float16* s0  = (_Float16*)take((size_t)NR * NC * sizeof(_Float16));
  _Float16* s1  = (_Float16*)take((size_t)NR * NC * sizeof(_Float16));
  _Float16* a0h = (_Float16*)take((size_t)NR * KPAD * sizeof(_Float16));
  _Float16* a1h = (_Float16*)take((size_t)NR * KPAD * sizeof(_Float16));
  _Float16* bh  = (_Float16*)take((size_t)NC * KPAD * sizeof(_Float16));
  float* u0     = (float*)take((size_t)NR * sizeof(float));
  float* u1     = (float*)take((size_t)NR * sizeof(float));
  float* v0     = (float*)take((size_t)NC * sizeof(float));
  float* v1     = (float*)take((size_t)NC * sizeof(float));
  float* lse0   = (float*)take((size_t)NR * sizeof(float));
  float* lse1   = (float*)take((size_t)NR * sizeof(float));
  float* colsum = (float*)take((size_t)NC * sizeof(float));

  const int rowBlocks  = NR / 8;               // 6314 (wave per row)
  const int gemmBlocks = (NC / 128) * MBLK;    // 16 * 395 = 6320
  const int colBlocks  = (NR + 31) / 32;       // 1579

  init_state<<<8, 256, 0, stream>>>(v0, v1, colsum, out);

  norm_rows<<<(NR + 7) / 8, 256, 0, stream>>>(nf0, a0h, NR);
  norm_rows<<<(NR + 7) / 8, 256, 0, stream>>>(nf1, a1h, NR);
  norm_rows<<<(NC + 7) / 8, 256, 0, stream>>>(cen, bh, NC);

  gemm_score<<<gemmBlocks, 256, 0, stream>>>(a0h, bh, s0);
  gemm_score<<<gemmBlocks, 256, 0, stream>>>(a1h, bh, s1);

  for (int it = 0; it < 10; ++it) {
    sinkhorn_row<<<rowBlocks, 256, 0, stream>>>(s0, v0, u0);
    sinkhorn_col<<<colBlocks, 256, 0, stream>>>(s0, u0, colsum);
    finish_v<<<8, 256, 0, stream>>>(colsum, v0);
  }
  for (int it = 0; it < 10; ++it) {
    sinkhorn_row<<<rowBlocks, 256, 0, stream>>>(s1, v1, u1);
    sinkhorn_col<<<colBlocks, 256, 0, stream>>>(s1, u1, colsum);
    finish_v<<<8, 256, 0, stream>>>(colsum, v1);
  }

  row_lse<<<rowBlocks, 256, 0, stream>>>(s0, lse0);
  row_lse<<<rowBlocks, 256, 0, stream>>>(s1, lse1);

  final_loss<<<colBlocks, 256, 0, stream>>>(s0, s1, u0, v0, u1, v1, lse0, lse1, out);
}
